// GTMEstimator_74706661147263
// MI455X (gfx1250) — matrix operand order repaired and checked by reading, not compiled, not run on hardware
//
#include <hip/hip_runtime.h>
#include <hip/hip_bf16.h>

typedef __attribute__((ext_vector_type(16))) _Float16 v16h;
typedef __attribute__((ext_vector_type(8)))  _Float16 v8h;
typedef __attribute__((ext_vector_type(4)))  _Float16 v4h;
typedef __attribute__((ext_vector_type(8)))  float    v8f;
typedef __attribute__((ext_vector_type(4)))  float    v4f;

#define KPTS       10000
#define KPAD       10240     // padded to 640 k-tiles so 8 waves x 2 tiles x 40 iters is exact
#define KTILES_PAD 640
#define DFEAT      512
#define NBATCH     8192
#define WG_COLS    32        // batch columns per workgroup (2 x 16-col WMMA tiles)
#define WAVES      8         // waves per workgroup (wave32)

// ---------------------------------------------------------------------------
// Kernel 1a: yg = grid @ W.T + b  (f16), and c_k = -beta/2 * ||yg_k||^2 (f32)
// One wave per grid point k. Pad rows (k >= 10000): yg = 0, c = -inf, so they
// contribute exp(-inf) = 0 to the softmax.
// ---------------------------------------------------------------------------
__global__ __launch_bounds__(256) void gtm_prep_yg(
    const float* __restrict__ grid, const float* __restrict__ W,
    const float* __restrict__ bias, const float* __restrict__ betta,
    _Float16* __restrict__ ygh, float* __restrict__ crow)
{
    const int wave = threadIdx.x >> 5;
    const int lane = threadIdx.x & 31;
    const int k    = blockIdx.x * WAVES + wave;   // k < KPAD

    if (k < KPTS) {
        const float g0 = grid[2 * k];
        const float g1 = grid[2 * k + 1];
        float sq = 0.f;
        #pragma unroll
        for (int i = 0; i < DFEAT / 32; ++i) {
            const int d = lane + 32 * i;
            const float v = fmaf(g0, W[2 * d], fmaf(g1, W[2 * d + 1], bias[d]));
            ygh[(size_t)k * DFEAT + d] = (_Float16)v;
            sq = fmaf(v, v, sq);
        }
        #pragma unroll
        for (int off = 16; off > 0; off >>= 1) sq += __shfl_xor(sq, off, 32);
        if (lane == 0) crow[k] = -0.5f * betta[0] * sq;
    } else {
        #pragma unroll
        for (int i = 0; i < DFEAT / 32; ++i)
            ygh[(size_t)k * DFEAT + lane + 32 * i] = (_Float16)0.f;
        if (lane == 0) crow[k] = -__builtin_inff();
    }
}

// ---------------------------------------------------------------------------
// Kernel 1b: X (f32) -> Xh (f16), vectorized 4 elements per thread
// ---------------------------------------------------------------------------
__global__ __launch_bounds__(256) void gtm_prep_x(
    const float* __restrict__ X, _Float16* __restrict__ Xh)
{
    const size_t i = ((size_t)blockIdx.x * 256 + threadIdx.x) * 4;
    v4f x = *(const v4f*)(X + i);
    v4h h;
    h[0] = (_Float16)x[0]; h[1] = (_Float16)x[1];
    h[2] = (_Float16)x[2]; h[3] = (_Float16)x[3];
    *(v4h*)(Xh + i) = h;
}

// ---------------------------------------------------------------------------
// Kernel 2: fused GEMM (f16 WMMA) + online softmax over the K (grid) axis +
// weighted reduction with grid coords. One WG per 32 batch columns. Each wave
// processes a 32-row x 32-col macro-tile per iteration (2 A frags x 2 B frags
// -> 4 WMMAs, every fragment reused twice: 1 fragment load per WMMA).
// ---------------------------------------------------------------------------
__global__ __launch_bounds__(256) void gtm_main(
    const _Float16* __restrict__ ygh, const _Float16* __restrict__ Xh,
    const float* __restrict__ crow, const float* __restrict__ grid,
    const float* __restrict__ betta, float* __restrict__ out)
{
    __shared__ float sm[WAVES * 64], ss[WAVES * 64], sa0[WAVES * 64], sa1[WAVES * 64];

    const int wave   = threadIdx.x >> 5;
    const int lane   = threadIdx.x & 31;
    const int lane16 = lane & 15;
    const int hf     = lane >> 4;          // lane half: selects K/M sub-block per ISA layout
    const int n0     = blockIdx.x * WG_COLS;
    const float beta = betta[0];

    // B fragment base pointers: column n = n0 + t*16 + lane16, K-half = hf*16.
    // X rows are contiguous along D, so each 32-deep B fragment is one 32B load.
    const _Float16* bptr0 = Xh + (size_t)(n0 + lane16) * DFEAT + hf * 16;
    const _Float16* bptr1 = Xh + (size_t)(n0 + 16 + lane16) * DFEAT + hf * 16;

    // Online-softmax state, one per 16-col tile (this lane's column).
    float m0 = -__builtin_inff(), s0 = 0.f, a00 = 0.f, a01 = 0.f;
    float m1 = -__builtin_inff(), s1 = 0.f, a10 = 0.f, a11 = 0.f;

    // 40 iterations: each covers 16 consecutive k-tiles (8 waves x 2 tiles).
    for (int t = 0; t < KTILES_PAD / (WAVES * 2); ++t) {
        const int kbase = (t * 16 + wave * 2) * 16;   // 32 rows: kbase .. kbase+31
        const _Float16* aptr0 = ygh + (size_t)(kbase + lane16) * DFEAT + hf * 16;
        const _Float16* aptr1 = aptr0 + (size_t)16 * DFEAT;

        v8f acc00 = {}, acc01 = {}, acc10 = {}, acc11 = {};
        #pragma unroll 2
        for (int d0 = 0; d0 < DFEAT; d0 += 32) {
            union { v16h v; v8h h[2]; } A0, A1;
            A0.h[0] = *(const v8h*)(aptr0 + d0);
            A0.h[1] = *(const v8h*)(aptr0 + d0 + 8);
            A1.h[0] = *(const v8h*)(aptr1 + d0);
            A1.h[1] = *(const v8h*)(aptr1 + d0 + 8);
            v16h B0 = *(const v16h*)(bptr0 + d0);
            v16h B1 = *(const v16h*)(bptr1 + d0);
            acc00 = __builtin_amdgcn_wmma_f32_16x16x32_f16(false, A0.v, false, B0,
                                                           (short)0, acc00, false, false);
            acc01 = __builtin_amdgcn_wmma_f32_16x16x32_f16(false, A0.v, false, B1,
                                                           (short)0, acc01, false, false);
            acc10 = __builtin_amdgcn_wmma_f32_16x16x32_f16(false, A1.v, false, B0,
                                                           (short)0, acc10, false, false);
            acc11 = __builtin_amdgcn_wmma_f32_16x16x32_f16(false, A1.v, false, B1,
                                                           (short)0, acc11, false, false);
        }

        // This lane holds rows {rbase..rbase+7} (k-tile 0) and {rbase+16..+23}
        // (k-tile 1) of both C tiles; column = lane16 within each n-tile.
        const int rbase = kbase + hf * 8;
        float l0[16], l1[16], gx[16], gy[16];
        #pragma unroll
        for (int r = 0; r < 16; ++r) {
            const int row = rbase + ((r < 8) ? r : (r + 8));
            const float c = crow[row];                       // padded, safe to KPAD
            const int rg  = (row < KPTS) ? row : (KPTS - 1); // clamp: p==0 for pad rows
            gx[r] = grid[2 * rg];
            gy[r] = grid[2 * rg + 1];
            const float v0 = (r < 8) ? acc00[r] : acc10[r - 8];
            const float v1 = (r < 8) ? acc01[r] : acc11[r - 8];
            l0[r] = fmaf(beta, v0, c);
            l1[r] = fmaf(beta, v1, c);
        }

        // tile 0 online-softmax update (16 logits)
        float tm = l0[0];
        #pragma unroll
        for (int r = 1; r < 16; ++r) tm = fmaxf(tm, l0[r]);
        float nm = fmaxf(m0, tm);
        float sc = __expf(m0 - nm);
        s0 *= sc; a00 *= sc; a01 *= sc; m0 = nm;
        #pragma unroll
        for (int r = 0; r < 16; ++r) {
            const float p = __expf(l0[r] - nm);
            s0 += p;
            a00 = fmaf(p, gx[r], a00);
            a01 = fmaf(p, gy[r], a01);
        }

        // tile 1 online-softmax update (16 logits)
        tm = l1[0];
        #pragma unroll
        for (int r = 1; r < 16; ++r) tm = fmaxf(tm, l1[r]);
        nm = fmaxf(m1, tm);
        sc = __expf(m1 - nm);
        s1 *= sc; a10 *= sc; a11 *= sc; m1 = nm;
        #pragma unroll
        for (int r = 0; r < 16; ++r) {
            const float p = __expf(l1[r] - nm);
            s1 += p;
            a10 = fmaf(p, gx[r], a10);
            a11 = fmaf(p, gy[r], a11);
        }
    }

    // Publish partial states: slot = wave*64 + tile*32 + (hf*16 + lane16)
    const int base = wave * 64 + lane;
    sm[base]      = m0; ss[base]      = s0; sa0[base]      = a00; sa1[base]      = a01;
    sm[base + 32] = m1; ss[base + 32] = s1; sa0[base + 32] = a10; sa1[base + 32] = a11;
    __syncthreads();

    // Merge 16 partials per column (8 waves x 2 lane-halves) via log-sum-exp combine.
    if (threadIdx.x < WG_COLS) {
        const int t  = threadIdx.x >> 4;
        const int nc = threadIdx.x & 15;
        float M = -__builtin_inff(), S = 0.f, A0 = 0.f, A1 = 0.f;
        #pragma unroll
        for (int w = 0; w < WAVES; ++w) {
            #pragma unroll
            for (int h = 0; h < 2; ++h) {
                const int slot = w * 64 + t * 32 + h * 16 + nc;
                const float m2 = sm[slot];
                const float nm = fmaxf(M, m2);
                const float e1 = __expf(M - nm);
                const float e2 = __expf(m2 - nm);
                S  = S  * e1 + ss[slot]  * e2;
                A0 = A0 * e1 + sa0[slot] * e2;
                A1 = A1 * e1 + sa1[slot] * e2;
                M  = nm;
            }
        }
        const int n = n0 + threadIdx.x;
        const float inv = 1.f / S;
        out[2 * n]     = A0 * inv;
        out[2 * n + 1] = A1 * inv;
    }
}

// ---------------------------------------------------------------------------
// Launch: inputs in setup_inputs() order: X, W, b, betta, grid
// ---------------------------------------------------------------------------
extern "C" void kernel_launch(void* const* d_in, const int* in_sizes, int n_in,
                              void* d_out, int out_size, void* d_ws, size_t ws_size,
                              hipStream_t stream) {
    const float* X     = (const float*)d_in[0];
    const float* W     = (const float*)d_in[1];
    const float* b     = (const float*)d_in[2];
    const float* betta = (const float*)d_in[3];
    const float* grid  = (const float*)d_in[4];
    float* out = (float*)d_out;

    // Workspace layout (all offsets 32B-aligned):
    //   ygh : 10240*512 f16 = 10,485,760 B  (rows 10000..10239 are zero padding)
    //   crow: 10240 f32     =     40,960 B  (pad rows = -inf)
    //   Xh  : 8192*512 f16  =  8,388,608 B  (total ~18.9 MB)
    char* ws = (char*)d_ws;
    _Float16* ygh  = (_Float16*)(ws);
    float*    crow = (float*)(ws + 10485760);
    _Float16* Xh   = (_Float16*)(ws + 10526720);

    gtm_prep_yg<<<KPAD / WAVES, 256, 0, stream>>>(grid, W, b, betta, ygh, crow);
    gtm_prep_x<<<(NBATCH * DFEAT) / (256 * 4), 256, 0, stream>>>(X, Xh);
    gtm_main<<<NBATCH / WG_COLS, 256, 0, stream>>>(ygh, Xh, crow, grid, betta, out);
}
